// ProjRNN_14955076124988
// MI455X (gfx1250) — matrix operand order repaired and checked by reading, not compiled, not run on hardware
//
#include <hip/hip_runtime.h>
#include <hip/hip_bf16.h>

// ProjRNN (hyper-GRU) for MI455X / gfx1250, wave32 + WMMA f16->f32.
//
// Phases:
//  1) hypergen:  w_ih/w_hh (f16) + b_ih/b_hh (f32) from base_proj  (BW-bound, 805MB)
//  2) gi GEMM:   [32768x512] x [512x1536] via v_wmma_f32_16x16x32_f16 (51.5 GFLOP),
//                software-pipelined K loop (loads for k+1 overlap WMMAs of k)
//  3) scan:      256 x fused step kernel: gh GEMM (3 gates in registers) + GRU gate
//                math on the WMMA accumulators; h16 ping-pongs between two buffers.

typedef __attribute__((ext_vector_type(16))) _Float16 v16h;
typedef __attribute__((ext_vector_type(8)))  float    v8f;

#define C_DIM 128
#define D_DIM 512
#define U_DIM 512
#define G_DIM 1536           // 3*U
#define B_DIM 128
#define T_DIM 256
#define M_DIM (B_DIM * T_DIM) // 32768 rows of the batched input projection

// ---------------- fragment loaders (ISA 7.12.2 layouts, wave32) ----------------

// A-matrix 16x32 f16 from an f32 row-major source (cvt on load).
__device__ inline v16h load_a_frag_f32(const float* __restrict__ A, int lda, int lane) {
    const int l  = lane & 15;
    const int hi = lane >> 4;
    const float* p = A + (size_t)l * lda + hi * 8;
    v16h a;
#pragma unroll
    for (int i = 0; i < 8; ++i) a[i]     = (_Float16)p[i];
#pragma unroll
    for (int i = 0; i < 8; ++i) a[i + 8] = (_Float16)p[16 + i];
    return a;
}

__device__ inline v16h load_a_frag_f16(const _Float16* __restrict__ A, int lda, int lane) {
    const int l  = lane & 15;
    const int hi = lane >> 4;
    const _Float16* p = A + (size_t)l * lda + hi * 8;
    v16h a;
#pragma unroll
    for (int i = 0; i < 8; ++i) a[i]     = p[i];
#pragma unroll
    for (int i = 0; i < 8; ++i) a[i + 8] = p[16 + i];
    return a;
}

// B-matrix 32x16 (KxN): B[k][n] = W[(g0+n)*ldw + k0 + k], W row-major [G,K].
__device__ inline v16h load_b_frag_f16(const _Float16* __restrict__ W, int ldw, int lane) {
    const int l  = lane & 15;
    const int hi = lane >> 4;
    const _Float16* p = W + (size_t)l * ldw + hi * 8;
    v16h b;
#pragma unroll
    for (int i = 0; i < 8; ++i) b[i]     = p[i];
#pragma unroll
    for (int i = 0; i < 8; ++i) b[i + 8] = p[16 + i];
    return b;
}

__device__ inline float sigmoidf_fast(float x) {
    return 1.f / (1.f + __expf(-x));
}

// ---------------- 1) hyper-network weight generation ----------------

__global__ __launch_bounds__(256) void hypergen_kernel(
    const float* __restrict__ bp,
    const float* __restrict__ Wih_w, const float* __restrict__ Wih_b,
    const float* __restrict__ Whh_w, const float* __restrict__ Whh_b,
    const float* __restrict__ bih_w, const float* __restrict__ bih_b,
    const float* __restrict__ bhh_w, const float* __restrict__ bhh_b,
    _Float16* __restrict__ w_ih, _Float16* __restrict__ w_hh,
    float* __restrict__ b_ih, float* __restrict__ b_hh)
{
    __shared__ float sbp[C_DIM];
    if (threadIdx.x < C_DIM) sbp[threadIdx.x] = bp[threadIdx.x];
    __syncthreads();

    const int r = blockIdx.x * 256 + threadIdx.x;     // grid sized exactly to G*D
    const float* p0 = Wih_w + (size_t)r * C_DIM;
    const float* p1 = Whh_w + (size_t)r * C_DIM;
    float acc0 = 0.f, acc1 = 0.f;
#pragma unroll 8
    for (int c = 0; c < C_DIM; c += 4) {
        const float4 a0 = *(const float4*)(p0 + c);
        const float4 a1 = *(const float4*)(p1 + c);
        acc0 += a0.x * sbp[c] + a0.y * sbp[c + 1] + a0.z * sbp[c + 2] + a0.w * sbp[c + 3];
        acc1 += a1.x * sbp[c] + a1.y * sbp[c + 1] + a1.z * sbp[c + 2] + a1.w * sbp[c + 3];
    }
    w_ih[r] = (_Float16)(acc0 + Wih_b[r]);
    w_hh[r] = (_Float16)(acc1 + Whh_b[r]);

    if (r < G_DIM) {
        const float* q0 = bih_w + (size_t)r * C_DIM;
        const float* q1 = bhh_w + (size_t)r * C_DIM;
        float s0 = 0.f, s1 = 0.f;
#pragma unroll 8
        for (int c = 0; c < C_DIM; ++c) { s0 += q0[c] * sbp[c]; s1 += q1[c] * sbp[c]; }
        b_ih[r] = s0 + bih_b[r];
        b_hh[r] = s1 + bhh_b[r];
    }
}

// ---------------- 2) batched input projection: gi = X @ w_ih^T + b_ih ----------------
// grid (M/128, G/64), block 256 = 8 waves; each wave -> 16x64 strip, pipelined K loop.

__global__ __launch_bounds__(256) void gi_gemm_kernel(
    const float* __restrict__ X,        // [M_DIM, D_DIM] f32
    const _Float16* __restrict__ Wih,   // [G_DIM, D_DIM] f16
    const float* __restrict__ b_ih,     // [G_DIM]
    float* __restrict__ gi)             // [M_DIM, G_DIM] f32
{
    const int lane = threadIdx.x & 31;
    const int wave = threadIdx.x >> 5;
    const int m0 = blockIdx.x * 128 + wave * 16;
    const int g0 = blockIdx.y * 64;

    const float*    Abase = X   + (size_t)m0 * D_DIM;
    const _Float16* Bbase = Wih + (size_t)g0 * D_DIM;

    v8f acc[4] = {};
    // prologue: fragments for k0 = 0
    v16h a_cur = load_a_frag_f32(Abase, D_DIM, lane);
    v16h b_cur[4];
#pragma unroll
    for (int j = 0; j < 4; ++j)
        b_cur[j] = load_b_frag_f16(Bbase + (size_t)(16 * j) * D_DIM, D_DIM, lane);

#pragma unroll
    for (int k0 = 32; k0 < D_DIM; k0 += 32) {
        // issue loads for this k0 before consuming the previous fragments
        v16h a_nxt = load_a_frag_f32(Abase + k0, D_DIM, lane);
        v16h b_nxt[4];
#pragma unroll
        for (int j = 0; j < 4; ++j)
            b_nxt[j] = load_b_frag_f16(Bbase + (size_t)(16 * j) * D_DIM + k0, D_DIM, lane);
#pragma unroll
        for (int j = 0; j < 4; ++j)
            acc[j] = __builtin_amdgcn_wmma_f32_16x16x32_f16(
                false, a_cur, false, b_cur[j], (short)0, acc[j], false, false);
        a_cur = a_nxt;
#pragma unroll
        for (int j = 0; j < 4; ++j) b_cur[j] = b_nxt[j];
    }
#pragma unroll
    for (int j = 0; j < 4; ++j)
        acc[j] = __builtin_amdgcn_wmma_f32_16x16x32_f16(
            false, a_cur, false, b_cur[j], (short)0, acc[j], false, false);

    const int l  = lane & 15;
    const int hi = lane >> 4;
#pragma unroll
    for (int j = 0; j < 4; ++j) {
        const int g = g0 + 16 * j + l;
        const float bias = b_ih[g];
#pragma unroll
        for (int i = 0; i < 8; ++i)
            gi[(size_t)(m0 + hi * 8 + i) * G_DIM + g] = acc[j][i] + bias;
    }
}

// ---------------- 3) fused recurrent step: gh GEMM (3 gates) + GRU update ----------------
// grid (U/32) = 16 blocks, block 256 = 8 waves; wave -> 16 batch rows x 32 u-cols,
// 3 gate accumulators kept in registers; gh never touches memory.

__global__ __launch_bounds__(256) void step_kernel(
    const _Float16* __restrict__ Hin,   // [B_DIM, U_DIM] f16 (state at t)
    const _Float16* __restrict__ Whh,   // [G_DIM, U_DIM] f16
    const float* __restrict__ bhh,      // [G_DIM]
    const float* __restrict__ gi,       // [M_DIM, G_DIM] f32 (b_ih already folded in)
    float* __restrict__ h32,            // [B_DIM, U_DIM] f32 (per-element private)
    _Float16* __restrict__ Hout,        // [B_DIM, U_DIM] f16 (state at t+1)
    float* __restrict__ out,            // [B_DIM, T_DIM, U_DIM]
    int t)
{
    const int lane = threadIdx.x & 31;
    const int wave = threadIdx.x >> 5;
    const int b0 = wave * 16;                 // 8 waves cover all 128 batch rows
    const int u0 = blockIdx.x * 32;           // 16 blocks cover all 512 hidden cols

    const _Float16* Abase = Hin + (size_t)b0 * U_DIM;

    v8f acc[3][2] = {};                       // [gate r/z/n][col tile]
    v16h a_cur = load_a_frag_f16(Abase, U_DIM, lane);
    v16h b_cur[3][2];
#pragma unroll
    for (int gq = 0; gq < 3; ++gq)
#pragma unroll
        for (int ct = 0; ct < 2; ++ct)
            b_cur[gq][ct] = load_b_frag_f16(
                Whh + (size_t)(gq * U_DIM + u0 + 16 * ct) * U_DIM, U_DIM, lane);

#pragma unroll
    for (int k0 = 32; k0 < U_DIM; k0 += 32) {
        v16h a_nxt = load_a_frag_f16(Abase + k0, U_DIM, lane);
        v16h b_nxt[3][2];
#pragma unroll
        for (int gq = 0; gq < 3; ++gq)
#pragma unroll
            for (int ct = 0; ct < 2; ++ct)
                b_nxt[gq][ct] = load_b_frag_f16(
                    Whh + (size_t)(gq * U_DIM + u0 + 16 * ct) * U_DIM + k0, U_DIM, lane);
#pragma unroll
        for (int gq = 0; gq < 3; ++gq)
#pragma unroll
            for (int ct = 0; ct < 2; ++ct)
                acc[gq][ct] = __builtin_amdgcn_wmma_f32_16x16x32_f16(
                    false, a_cur, false, b_cur[gq][ct], (short)0, acc[gq][ct], false, false);
        a_cur = a_nxt;
#pragma unroll
        for (int gq = 0; gq < 3; ++gq)
#pragma unroll
            for (int ct = 0; ct < 2; ++ct) b_cur[gq][ct] = b_nxt[gq][ct];
    }
#pragma unroll
    for (int gq = 0; gq < 3; ++gq)
#pragma unroll
        for (int ct = 0; ct < 2; ++ct)
            acc[gq][ct] = __builtin_amdgcn_wmma_f32_16x16x32_f16(
                false, a_cur, false, b_cur[gq][ct], (short)0, acc[gq][ct], false, false);

    // GRU gate math directly on the accumulators.
    const int l  = lane & 15;
    const int hi = lane >> 4;
#pragma unroll
    for (int ct = 0; ct < 2; ++ct) {
        const int u = u0 + 16 * ct + l;
        const float bh_r = bhh[u];
        const float bh_z = bhh[U_DIM + u];
        const float bh_n = bhh[2 * U_DIM + u];
#pragma unroll
        for (int i = 0; i < 8; ++i) {
            const int b = b0 + hi * 8 + i;
            const size_t m = (size_t)b * T_DIM + t;
            const float* girow = gi + m * G_DIM;
            const float r = sigmoidf_fast(girow[u]             + acc[0][ct][i] + bh_r);
            const float z = sigmoidf_fast(girow[U_DIM + u]     + acc[1][ct][i] + bh_z);
            const float n = tanhf(girow[2 * U_DIM + u] + r * (acc[2][ct][i] + bh_n));
            const size_t hidx = (size_t)b * U_DIM + u;
            const float hp = h32[hidx];
            const float hn = (1.f - z) * n + z * hp;
            h32[hidx]  = hn;
            Hout[hidx] = (_Float16)hn;
            out[m * U_DIM + u] = hn;
        }
    }
}

__global__ __launch_bounds__(256) void hinit_kernel(float* __restrict__ h32,
                                                    _Float16* __restrict__ h16)
{
    const int idx = blockIdx.x * 256 + threadIdx.x;
    h32[idx] = 0.f;
    h16[idx] = (_Float16)0.f;
}

// ---------------- launch ----------------

extern "C" void kernel_launch(void* const* d_in, const int* in_sizes, int n_in,
                              void* d_out, int out_size, void* d_ws, size_t ws_size,
                              hipStream_t stream) {
    (void)in_sizes; (void)n_in; (void)out_size; (void)ws_size;
    const float* input = (const float*)d_in[0];
    const float* bp    = (const float*)d_in[1];
    const float* Wih_w = (const float*)d_in[2];
    const float* Wih_b = (const float*)d_in[3];
    const float* Whh_w = (const float*)d_in[4];
    const float* Whh_b = (const float*)d_in[5];
    const float* bih_w = (const float*)d_in[6];
    const float* bih_b = (const float*)d_in[7];
    const float* bhh_w = (const float*)d_in[8];
    const float* bhh_b = (const float*)d_in[9];
    float* out = (float*)d_out;

    // workspace layout (256B-aligned blocks)
    char* ws = (char*)d_ws;
    _Float16* w_ih = (_Float16*)(ws + 0x000000);            // 1,572,864 B
    _Float16* w_hh = (_Float16*)(ws + 0x180000);            // 1,572,864 B
    float*    b_ih = (float*)   (ws + 0x300000);            //     6,144 B
    float*    b_hh = (float*)   (ws + 0x302000);            //     6,144 B
    float*    h32  = (float*)   (ws + 0x304000);            //   262,144 B
    _Float16* h16a = (_Float16*)(ws + 0x344000);            //   131,072 B
    _Float16* h16b = (_Float16*)(ws + 0x364000);            //   131,072 B
    float*    gi   = (float*)   (ws + 0x440000);            // 201,326,592 B

    // 1) generate GRU weights from base_proj (BW-bound, ~805MB reads)
    hypergen_kernel<<<(G_DIM * D_DIM) / 256, 256, 0, stream>>>(
        bp, Wih_w, Wih_b, Whh_w, Whh_b, bih_w, bih_b, bhh_w, bhh_b,
        w_ih, w_hh, b_ih, b_hh);

    // 2) time-batched input projection (WMMA, fully parallel)
    gi_gemm_kernel<<<dim3(M_DIM / 128, G_DIM / 64), 256, 0, stream>>>(
        input, w_ih, b_ih, gi);

    // 3) sequential scan: one fused kernel per step, h16 ping-pong
    hinit_kernel<<<(B_DIM * U_DIM) / 256, 256, 0, stream>>>(h32, h16a);
    _Float16* hA = h16a;
    _Float16* hB = h16b;
    for (int t = 0; t < T_DIM; ++t) {
        step_kernel<<<U_DIM / 32, 256, 0, stream>>>(hA, w_hh, b_hh, gi, h32, hB, out, t);
        _Float16* tmp = hA; hA = hB; hB = tmp;
    }
}
